// EdgeReweighting_69389491634806
// MI455X (gfx1250) — compile-verified, matrix operand order checked by reading, not run on hardware
//
#include <hip/hip_runtime.h>
#include <hip/hip_bf16.h>

typedef __attribute__((ext_vector_type(16))) _Float16 v16h;
typedef __attribute__((ext_vector_type(8)))  float    v8f;

#define D_FEAT   64
#define D_IN     192          // 3 * 64 concatenated features
#define D_HID    128
#define KBLK     6            // 192 / 32
#define NTILE    8            // 128 / 16
#define W1_HALFS (D_IN * D_HID)   // 24576 halves per matrix
#define LDS_BYTES (2 * W1_HALFS * 2) // 98304 B: gate + score w1 as f16

// ---------- order-preserving float <-> uint for atomic segment max ----------
__device__ __forceinline__ unsigned enc_f32(float x) {
  unsigned u = __float_as_uint(x);
  return (u & 0x80000000u) ? ~u : (u | 0x80000000u);
}
__device__ __forceinline__ float dec_f32(unsigned u) {
  return (u & 0x80000000u) ? __uint_as_float(u & 0x7fffffffu)
                           : __uint_as_float(~u);
}

// Build an A fragment (16x32 f16, wave32 layout) for one 32-wide K block of a
// 64-float feature row.  Lane group 0 (lanes 0-15) holds K = k0+{0..7,16..23},
// lane group 1 (lanes 16-31) holds K = k0+{8..15,24..31}; caller folds the +8
// into k0 via laneoff.
__device__ __forceinline__ v16h make_frag(const float* __restrict__ row, int k0) {
  float4 f0 = *(const float4*)(row + k0);
  float4 f1 = *(const float4*)(row + k0 + 4);
  float4 f2 = *(const float4*)(row + k0 + 16);
  float4 f3 = *(const float4*)(row + k0 + 20);
  v16h a;
  a[0]  = (_Float16)f0.x; a[1]  = (_Float16)f0.y; a[2]  = (_Float16)f0.z; a[3]  = (_Float16)f0.w;
  a[4]  = (_Float16)f1.x; a[5]  = (_Float16)f1.y; a[6]  = (_Float16)f1.z; a[7]  = (_Float16)f1.w;
  a[8]  = (_Float16)f2.x; a[9]  = (_Float16)f2.y; a[10] = (_Float16)f2.z; a[11] = (_Float16)f2.w;
  a[12] = (_Float16)f3.x; a[13] = (_Float16)f3.y; a[14] = (_Float16)f3.z; a[15] = (_Float16)f3.w;
  return a;
}

// One 192->128->1 MLP for 16 edges using 48 WMMAs; writes 16 pre-activation
// scalars (one per edge/row) into tmp16.
__device__ __forceinline__ void mlp16(const v16h* __restrict__ af,
                                      const _Float16* __restrict__ bmat,
                                      const float* __restrict__ b1,
                                      const float* __restrict__ w2,
                                      float b2, int lane,
                                      volatile float* tmp16) {
  const _Float16* bbase = bmat + lane * 16;   // per-lane fragment base
  float s[8] = {0.f, 0.f, 0.f, 0.f, 0.f, 0.f, 0.f, 0.f};
  const int nloc = lane & 15;
#pragma unroll
  for (int nt = 0; nt < NTILE; ++nt) {
    float bb = b1[nt * 16 + nloc];
    v8f acc = {bb, bb, bb, bb, bb, bb, bb, bb};
#pragma unroll
    for (int kb = 0; kb < KBLK; ++kb) {
      v16h bf = *(const v16h*)(bbase + (kb * NTILE + nt) * 512);
      acc = __builtin_amdgcn_wmma_f32_16x16x32_f16(
          false, af[kb], false, bf, (short)0, acc, false, false);
    }
    float wv = w2[nt * 16 + nloc];
#pragma unroll
    for (int r = 0; r < 8; ++r) s[r] += fmaxf(acc[r], 0.0f) * wv;
  }
  // reduce across each 16-lane half: lanes 0-15 carry rows r, 16-31 rows r+8
#pragma unroll
  for (int r = 0; r < 8; ++r) {
    float v = s[r];
    v += __shfl_xor(v, 1, 32);
    v += __shfl_xor(v, 2, 32);
    v += __shfl_xor(v, 4, 32);
    v += __shfl_xor(v, 8, 32);
    v += b2;
    if (lane == 0)       tmp16[r]     = v;
    else if (lane == 16) tmp16[r + 8] = v;
  }
}

__global__ void __launch_bounds__(256)
edge_wmma_kernel(const float* __restrict__ h, const float* __restrict__ e,
                 const float* __restrict__ q,
                 const int* __restrict__ src, const int* __restrict__ dst,
                 const int* __restrict__ eb,
                 const float* __restrict__ gw1, const float* __restrict__ gb1,
                 const float* __restrict__ gw2, const float* __restrict__ gb2,
                 const float* __restrict__ sw1, const float* __restrict__ sb1,
                 const float* __restrict__ sw2, const float* __restrict__ sb2,
                 float* __restrict__ raw, unsigned* __restrict__ maxenc, int E) {
  extern __shared__ _Float16 smem_w[];          // [gate w1 | score w1], swizzled
  __shared__ float tmpbuf[8][2][16];            // [wave][mlp][edge-in-tile]
  _Float16* wg = smem_w;
  _Float16* ws = smem_w + W1_HALFS;

  const int tid = threadIdx.x;
  // ---- stage + swizzle both w1 matrices into LDS as f16 fragments ----
  for (int idx = tid; idx < W1_HALFS; idx += 256) {
    int k = idx >> 7, n = idx & 127;
    int kb = k >> 5, r = k & 31;
    int grp  = (r >> 3) & 1;                         // lane half
    int elem = (r & 7) | (((r >> 4) & 1) << 3);      // slot within v16h
    int lane = grp * 16 + (n & 15);
    int nt   = n >> 4;
    int dsti = (kb * NTILE + nt) * 512 + lane * 16 + elem;
    wg[dsti] = (_Float16)gw1[idx];
    ws[dsti] = (_Float16)sw1[idx];
  }
  __syncthreads();

  const int lane = tid & 31;
  const int wave = tid >> 5;
  const int wavebase = blockIdx.x * 128 + wave * 16;
  const int m = lane & 15;
  const int laneoff = (lane & 16) >> 1;              // 0 or 8

  int eidx = wavebase + m;
  int eclmp = eidx < E ? eidx : (E - 1);
  int s_i = src[eclmp];
  int d_i = dst[eclmp];
  int b_i = eb[eclmp];

  const float* hs = h + (size_t)s_i * D_FEAT;
  const float* hd = h + (size_t)d_i * D_FEAT;
  const float* qr = q + (size_t)b_i * D_FEAT;
  const float* er = e + (size_t)eclmp * D_FEAT;

  v16h af[KBLK];
  af[0] = make_frag(hs, 0 + laneoff);
  af[1] = make_frag(hs, 32 + laneoff);
  af[2] = make_frag(hd, 0 + laneoff);
  af[3] = make_frag(hd, 32 + laneoff);

  // gate MLP: [h_i | h_j | q]
  af[4] = make_frag(qr, 0 + laneoff);
  af[5] = make_frag(qr, 32 + laneoff);
  mlp16(af, wg, gb1, gw2, gb2[0], lane, &tmpbuf[wave][0][0]);

  // score MLP: [h_i | h_j | e]
  af[4] = make_frag(er, 0 + laneoff);
  af[5] = make_frag(er, 32 + laneoff);
  mlp16(af, ws, sb1, sw2, sb2[0], lane, &tmpbuf[wave][1][0]);

  __builtin_amdgcn_wave_barrier();
  asm volatile("s_wait_dscnt 0" ::: "memory");

  if (lane < 16 && eidx < E) {
    float g  = tmpbuf[wave][0][m];
    float sc = tmpbuf[wave][1][m];
    float gate = 1.0f / (1.0f + __expf(-g));
    float rv = gate * sc;
    raw[eidx] = rv;
    atomicMax(&maxenc[d_i], enc_f32(rv));          // global_atomic_max_u32
  }
}

__global__ void zero_kernel(unsigned* __restrict__ p, int n) {
  int i = blockIdx.x * blockDim.x + threadIdx.x;
  if (i < n) p[i] = 0u;                            // 0 encodes "below all floats"
}

__global__ void exp_kernel(float* __restrict__ raw,
                           const unsigned* __restrict__ maxenc,
                           float* __restrict__ sum,
                           const int* __restrict__ dst, int E) {
  int i = blockIdx.x * blockDim.x + threadIdx.x;
  if (i < E) {
    int d = dst[i];
    float mx = dec_f32(maxenc[d]);
    float ex = __expf(raw[i] - mx);
    raw[i] = ex;
    unsafeAtomicAdd(&sum[d], ex);                  // global_atomic_add_f32
  }
}

__global__ void norm_kernel(const float* __restrict__ raw,
                            const float* __restrict__ sum,
                            const int* __restrict__ dst,
                            float* __restrict__ out, int E) {
  int i = blockIdx.x * blockDim.x + threadIdx.x;
  if (i < E) out[i] = raw[i] / fmaxf(sum[dst[i]], 1e-9f);
}

extern "C" void kernel_launch(void* const* d_in, const int* in_sizes, int n_in,
                              void* d_out, int out_size, void* d_ws, size_t ws_size,
                              hipStream_t stream) {
  const float* h   = (const float*)d_in[0];
  const float* e   = (const float*)d_in[1];
  const float* q   = (const float*)d_in[2];
  const int*   ei  = (const int*)d_in[3];
  const int*   eb  = (const int*)d_in[4];
  const float* gw1 = (const float*)d_in[5];
  const float* gb1 = (const float*)d_in[6];
  const float* gw2 = (const float*)d_in[7];
  const float* gb2 = (const float*)d_in[8];
  const float* sw1 = (const float*)d_in[9];
  const float* sb1 = (const float*)d_in[10];
  const float* sw2 = (const float*)d_in[11];
  const float* sb2 = (const float*)d_in[12];

  const int E = in_sizes[4];            // edge_batch length
  const int N = in_sizes[0] / D_FEAT;   // node count
  const int* src = ei;
  const int* dst = ei + E;

  float*    raw    = (float*)d_ws;
  unsigned* maxenc = (unsigned*)(raw + E);
  float*    sum    = (float*)(maxenc + N);   // contiguous after maxenc

  (void)hipFuncSetAttribute((const void*)edge_wmma_kernel,
                            hipFuncAttributeMaxDynamicSharedMemorySize,
                            LDS_BYTES);

  // zero maxenc (N u32) and sum (N f32, bit-zero == 0.0f) in one shot
  zero_kernel<<<(2 * N + 255) / 256, 256, 0, stream>>>(maxenc, 2 * N);

  edge_wmma_kernel<<<(E + 127) / 128, 256, LDS_BYTES, stream>>>(
      h, e, q, src, dst, eb, gw1, gb1, gw2, gb2, sw1, sb1, sw2, sb2,
      raw, maxenc, E);

  exp_kernel<<<(E + 255) / 256, 256, 0, stream>>>(raw, maxenc, sum, dst, E);

  norm_kernel<<<(E + 255) / 256, 256, 0, stream>>>(raw, sum, dst,
                                                   (float*)d_out, E);
}
